// BaselineGPTLanguageModel_26852135534687
// MI455X (gfx1250) — matrix operand order repaired and checked by reading, not compiled, not run on hardware
//
#include <hip/hip_runtime.h>

typedef unsigned short u16;
typedef __attribute__((ext_vector_type(16))) __bf16 v16bf;
typedef __attribute__((ext_vector_type(8)))  float  v8f;

#define GPT_B  4
#define GPT_T  1024
#define GPT_D  512
#define GPT_H  8
#define GPT_HS 64
#define GPT_L  6
#define GPT_FF 2048
#define GPT_V  50257
#define GPT_VP 50304
#define GPT_BT 4096

__device__ __forceinline__ u16 f2bf(float f) {
  unsigned u = __float_as_uint(f);
  u += 0x7FFFu + ((u >> 16) & 1u);   // round-to-nearest-even
  return (u16)(u >> 16);
}

__device__ __forceinline__ v16bf frag16(const u16* p0, const u16* p1) {
  union { v16bf v; uint4 q[2]; } f;
  f.q[0] = *(const uint4*)p0;
  f.q[1] = *(const uint4*)p1;
  return f.v;
}

__device__ __forceinline__ v8f wmma_bf16(v16bf a, v16bf b, v8f c) {
  return __builtin_amdgcn_wmma_f32_16x16x32_bf16(false, a, false, b, (short)0, c,
                                                 false, false);
}

// issue one 16B async global->LDS copy (per lane); tracked by ASYNCcnt
__device__ __forceinline__ void async_copy16(void* lds_ptr, const void* gptr) {
  unsigned lds = (unsigned)(size_t)lds_ptr;                 // LDS offset = addr[31:0]
  unsigned long long ga = (unsigned long long)(size_t)gptr;
  asm volatile("global_load_async_to_lds_b128 %0, %1, off"
               :: "v"(lds), "v"(ga) : "memory");
}

__device__ __forceinline__ void wait_async0() {
  asm volatile("s_wait_asynccnt 0x0" ::: "memory");
}

// ---------------------------------------------------------------- conversions
__global__ void conv_qkv_kernel(const float* __restrict__ w, u16* __restrict__ wt) {
  long long i = (long long)blockIdx.x * 256 + threadIdx.x;     // L*512*512
  if (i >= (long long)GPT_L * GPT_D * GPT_D) return;
  int d = i & 511; long long r = i >> 9;
  int n = (int)(r & 511); int l = (int)(r >> 9);
  int h = n >> 6, hs = n & 63;
  // src [L,H,D,HS] -> dst [L][H*HS][D]
  wt[i] = f2bf(w[(((size_t)l * GPT_H + h) * GPT_D + d) * GPT_HS + hs]);
}

__global__ void conv_plain_kernel(const float* __restrict__ w, u16* __restrict__ wt,
                                  int K, int N, int Np, long long total) {
  long long i = (long long)blockIdx.x * 256 + threadIdx.x;     // nm*Np*K
  if (i >= total) return;
  int kq = (int)(i % K); long long r = i / K;
  int n = (int)(r % Np); int mat = (int)(r / Np);
  float v = (n < N) ? w[((size_t)mat * K + kq) * N + n] : 0.f;
  wt[i] = f2bf(v);
}

// ---------------------------------------------------------------- embedding
__global__ void embed_kernel(const int* __restrict__ idx, const float* __restrict__ tok,
                             const float* __restrict__ pos, float* __restrict__ x) {
  int i = blockIdx.x * 256 + threadIdx.x;                      // BT*D
  if (i >= GPT_BT * GPT_D) return;
  int d = i & 511; int bt = i >> 9; int t = bt & 1023;
  x[i] = tok[(size_t)idx[bt] * GPT_D + d] + pos[t * GPT_D + d];
}

// ---------------------------------------------------------------- layernorm
__global__ __launch_bounds__(256) void ln_kernel(const float* __restrict__ x,
                                                 const float* __restrict__ g,
                                                 const float* __restrict__ bta,
                                                 u16* __restrict__ out) {
  __shared__ float red[16];
  const int row = blockIdx.x;
  const int tid = threadIdx.x;
  const float* xr = x + (size_t)row * GPT_D;
  float v0 = xr[tid], v1 = xr[tid + 256];
  float s = v0 + v1, sq = v0 * v0 + v1 * v1;
  for (int d = 1; d < 32; d <<= 1) {
    s  += __shfl_xor(s,  d, 32);
    sq += __shfl_xor(sq, d, 32);
  }
  int w = tid >> 5;
  if ((tid & 31) == 0) { red[w] = s; red[8 + w] = sq; }
  __syncthreads();
  float ts = 0.f, tsq = 0.f;
  for (int i = 0; i < 8; ++i) { ts += red[i]; tsq += red[8 + i]; }
  float mu = ts * (1.f / GPT_D);
  float var = tsq * (1.f / GPT_D) - mu * mu;
  float rs = rsqrtf(var + 1e-5f);
  out[(size_t)row * GPT_D + tid]       = f2bf((v0 - mu) * rs * g[tid] + bta[tid]);
  out[(size_t)row * GPT_D + tid + 256] = f2bf((v1 - mu) * rs * g[tid + 256] + bta[tid + 256]);
}

// ---------------------------------------------------------------- V transpose
__global__ void vtrans_kernel(const u16* __restrict__ v, u16* __restrict__ vt) {
  int i = blockIdx.x * 256 + threadIdx.x;                      // B*H*HS*T
  if (i >= GPT_B * GPT_H * GPT_HS * GPT_T) return;
  int t = i & 1023; int r = i >> 10;
  int hs = r & 63; r >>= 6;
  int h = r & 7; int b = r >> 3;
  vt[i] = v[(size_t)(b * GPT_T + t) * GPT_D + h * GPT_HS + hs];
}

// ---------------------------------------------------------------- GEMM (bf16 WMMA)
// C[M,N] = A[M,K] * Bt[N,K]^T.  M=4096, N%128==0, K%32==0.
// mode 0: bf16 store; 1: bf16 bias+relu; 2: f32 += acc+bias; 3: f32 store (n<nreal)
// Double-buffered LDS, tiles staged with global_load_async_to_lds_b128.
#define BM 128
#define BN 128
#define LDSA 40

__global__ __launch_bounds__(256) void gemm_bf16_kernel(
    const u16* __restrict__ A, const u16* __restrict__ Bt,
    const float* __restrict__ bias, void* __restrict__ C,
    int K, int ldc, int nreal, int mode) {
  __shared__ u16 As[2][BM * LDSA];
  __shared__ u16 Bs[2][BN * LDSA];
  const int tid = threadIdx.x;
  const int bm = blockIdx.y * BM;
  const int bn = blockIdx.x * BN;
  const int lane = tid & 31;
  const int wid  = tid >> 5;
  const int half = lane >> 4;
  const int lrow = lane & 15;
  const int kbA  = half * 8;
  const int wm = (wid & 3) * 32;
  const int wn = (wid >> 2) * 64;

  v8f acc[2][4];
#pragma unroll
  for (int i = 0; i < 2; ++i)
#pragma unroll
    for (int j = 0; j < 4; ++j)
#pragma unroll
      for (int e = 0; e < 8; ++e) acc[i][j][e] = 0.f;

  // precomputed per-thread copy coordinates (2 chunks per tile per matrix)
  const int r0 = tid >> 2, c0 = (tid & 3) << 3;            // chunk tid
  const int r1 = (tid + 256) >> 2, c1 = ((tid + 256) & 3) << 3;

  auto issue_tile = [&](int kk, int buf) {
    async_copy16(&As[buf][r0 * LDSA + c0], &A[(size_t)(bm + r0) * K + kk + c0]);
    async_copy16(&As[buf][r1 * LDSA + c1], &A[(size_t)(bm + r1) * K + kk + c1]);
    async_copy16(&Bs[buf][r0 * LDSA + c0], &Bt[(size_t)(bn + r0) * K + kk + c0]);
    async_copy16(&Bs[buf][r1 * LDSA + c1], &Bt[(size_t)(bn + r1) * K + kk + c1]);
  };

  issue_tile(0, 0);
  int buf = 0;
  for (int kk = 0; kk < K; kk += 32) {
    wait_async0();           // this wave's pending tile copies are in LDS
    __syncthreads();         // all waves' copies visible; prev buffer free
    if (kk + 32 < K) issue_tile(kk + 32, buf ^ 1);

    v16bf af[2], bfr[4];
#pragma unroll
    for (int i = 0; i < 2; ++i) {
      const u16* r = &As[buf][(wm + i * 16 + lrow) * LDSA];
      af[i] = frag16(r + kbA, r + kbA + 16);
    }
#pragma unroll
    for (int j = 0; j < 4; ++j) {
      const u16* r = &Bs[buf][(wn + j * 16 + lrow) * LDSA];
      bfr[j] = frag16(r + kbA, r + kbA + 16);
    }
#pragma unroll
    for (int i = 0; i < 2; ++i)
#pragma unroll
      for (int j = 0; j < 4; ++j)
        acc[i][j] = wmma_bf16(af[i], bfr[j], acc[i][j]);
    buf ^= 1;
  }

#pragma unroll
  for (int i = 0; i < 2; ++i)
#pragma unroll
    for (int j = 0; j < 4; ++j) {
      int mbase = bm + wm + i * 16 + half * 8;
      int n = bn + wn + j * 16 + lrow;
#pragma unroll
      for (int r = 0; r < 8; ++r) {
        float v = acc[i][j][r];
        size_t off = (size_t)(mbase + r) * (size_t)ldc + n;
        if (mode == 0) {
          ((u16*)C)[off] = f2bf(v);
        } else if (mode == 1) {
          float t = v + bias[n];
          ((u16*)C)[off] = f2bf(t > 0.f ? t : 0.f);
        } else if (mode == 2) {
          ((float*)C)[off] += v + bias[n];
        } else {
          if (n < nreal) ((float*)C)[off] = v;
        }
      }
    }
}

// ---------------------------------------------------------------- fused attention
// one block = one (b, h, 64-query tile); 4 waves, 16 query rows each.
#define ATS 72

__global__ __launch_bounds__(128) void attn_kernel(
    const u16* __restrict__ q, const u16* __restrict__ k,
    const u16* __restrict__ vt, u16* __restrict__ out) {
  __shared__ u16 Ks[64 * ATS];
  __shared__ u16 Vs[64 * ATS];
  __shared__ u16 Ps[64 * ATS];
  const int qt = blockIdx.x;
  const int h  = blockIdx.y;
  const int b  = blockIdx.z;
  const int tid = threadIdx.x;
  const int w = tid >> 5, lane = tid & 31;
  const int half = lane >> 4, lrow = lane & 15;
  const int kbA = half * 8;
  const int qr = qt * 64 + w * 16;

  v16bf aq[2];
  {
    const u16* qp = q + (size_t)(b * GPT_T + qr + lrow) * GPT_D + h * GPT_HS;
    aq[0] = frag16(qp + kbA, qp + kbA + 16);
    aq[1] = frag16(qp + 32 + kbA, qp + 32 + kbA + 16);
  }

  float mrow[8], lsum[8];
  v8f o[4];
#pragma unroll
  for (int r = 0; r < 8; ++r) { mrow[r] = -__builtin_inff(); lsum[r] = 0.f; }
#pragma unroll
  for (int j = 0; j < 4; ++j)
#pragma unroll
    for (int e = 0; e < 8; ++e) o[j][e] = 0.f;

  for (int st = 0; st <= qt; ++st) {
    const int s0 = st * 64;
    __syncthreads();                       // previous tile fully consumed
#pragma unroll
    for (int u = 0; u < 4; ++u) {
      int c = tid + u * 128;
      int row = c >> 3, col = (c & 7) << 3;
      async_copy16(&Ks[row * ATS + col],
                   &k[(size_t)(b * GPT_T + s0 + row) * GPT_D + h * GPT_HS + col]);
      async_copy16(&Vs[row * ATS + col],
                   &vt[((size_t)(b * GPT_H + h) * GPT_HS + row) * GPT_T + s0 + col]);
    }
    wait_async0();
    __syncthreads();

    v8f sc[4];
#pragma unroll
    for (int j = 0; j < 4; ++j) {
#pragma unroll
      for (int e = 0; e < 8; ++e) sc[j][e] = 0.f;
#pragma unroll
      for (int kt = 0; kt < 2; ++kt) {
        const u16* r = &Ks[(j * 16 + lrow) * ATS + kt * 32];
        v16bf bk = frag16(r + kbA, r + kbA + 16);
        sc[j] = wmma_bf16(aq[kt], bk, sc[j]);
      }
    }

#pragma unroll
    for (int r = 0; r < 8; ++r) {
      const int tq = qr + half * 8 + r;
      float mx = -__builtin_inff();
#pragma unroll
      for (int j = 0; j < 4; ++j) {
        int s = s0 + j * 16 + lrow;
        float v = (s <= tq) ? sc[j][r] * 0.125f : -__builtin_inff();
        sc[j][r] = v;
        mx = fmaxf(mx, v);
      }
      for (int d = 1; d < 16; d <<= 1) mx = fmaxf(mx, __shfl_xor(mx, d, 32));
      float mnew = fmaxf(mrow[r], mx);
      float scale = __expf(mrow[r] - mnew);
      mrow[r] = mnew;
      float ps = 0.f;
#pragma unroll
      for (int j = 0; j < 4; ++j) {
        float p = __expf(sc[j][r] - mnew);
        sc[j][r] = p;
        ps += p;
      }
      for (int d = 1; d < 16; d <<= 1) ps += __shfl_xor(ps, d, 32);
      lsum[r] = lsum[r] * scale + ps;
#pragma unroll
      for (int j = 0; j < 4; ++j) o[j][r] *= scale;
      const int ql = w * 16 + half * 8 + r;
#pragma unroll
      for (int j = 0; j < 4; ++j)
        Ps[ql * ATS + j * 16 + lrow] = f2bf(sc[j][r]);
    }

    // LDS ops from the same wave are processed in order: safe to re-read Ps.
    v16bf ap[2];
    {
      const u16* r = &Ps[(w * 16 + lrow) * ATS];
      ap[0] = frag16(r + kbA, r + kbA + 16);
      ap[1] = frag16(r + 32 + kbA, r + 32 + kbA + 16);
    }
#pragma unroll
    for (int j = 0; j < 4; ++j)
#pragma unroll
      for (int kt = 0; kt < 2; ++kt) {
        const u16* r = &Vs[(j * 16 + lrow) * ATS + kt * 32];
        v16bf bv = frag16(r + kbA, r + kbA + 16);
        o[j] = wmma_bf16(ap[kt], bv, o[j]);
      }
  }

#pragma unroll
  for (int j = 0; j < 4; ++j)
#pragma unroll
    for (int r = 0; r < 8; ++r) {
      float v = o[j][r] / lsum[r];
      out[(size_t)(b * GPT_T + qr + half * 8 + r) * GPT_D + h * GPT_HS + j * 16 + lrow] =
          f2bf(v);
    }
}

// ---------------------------------------------------------------- host
extern "C" void kernel_launch(void* const* d_in, const int* in_sizes, int n_in,
                              void* d_out, int out_size, void* d_ws, size_t ws_size,
                              hipStream_t stream) {
  (void)in_sizes; (void)n_in; (void)out_size; (void)ws_size;
  const int*   index = (const int*)  d_in[0];
  const float* tok   = (const float*)d_in[1];
  const float* pos   = (const float*)d_in[2];
  const float* Wq    = (const float*)d_in[3];
  const float* Wk    = (const float*)d_in[4];
  const float* Wv    = (const float*)d_in[5];
  const float* Wproj = (const float*)d_in[6];
  const float* bproj = (const float*)d_in[7];
  const float* ln1g  = (const float*)d_in[8];
  const float* ln1b  = (const float*)d_in[9];
  const float* ln2g  = (const float*)d_in[10];
  const float* ln2b  = (const float*)d_in[11];
  const float* W1    = (const float*)d_in[12];
  const float* b1    = (const float*)d_in[13];
  const float* W2    = (const float*)d_in[14];
  const float* b2    = (const float*)d_in[15];
  const float* lnfg  = (const float*)d_in[16];
  const float* lnfb  = (const float*)d_in[17];
  const float* Wlm   = (const float*)d_in[18];
  float* out = (float*)d_out;

  char* p = (char*)d_ws;
  auto take = [&](size_t bytes) -> char* {
    char* r = p;
    p += (bytes + 255) & ~(size_t)255;
    return r;
  };
  float* x   = (float*)take((size_t)GPT_BT * GPT_D * 4);
  u16* hb    = (u16*)take((size_t)GPT_BT * GPT_D * 2);
  u16* qb    = (u16*)take((size_t)GPT_BT * GPT_D * 2);
  u16* kb    = (u16*)take((size_t)GPT_BT * GPT_D * 2);
  u16* vb    = (u16*)take((size_t)GPT_BT * GPT_D * 2);
  u16* vtb   = (u16*)take((size_t)GPT_BT * GPT_D * 2);
  u16* ab    = (u16*)take((size_t)GPT_BT * GPT_D * 2);
  u16* ffm   = (u16*)take((size_t)GPT_BT * GPT_FF * 2);
  u16* wqt   = (u16*)take((size_t)GPT_L * GPT_D * GPT_D * 2);
  u16* wkt   = (u16*)take((size_t)GPT_L * GPT_D * GPT_D * 2);
  u16* wvt   = (u16*)take((size_t)GPT_L * GPT_D * GPT_D * 2);
  u16* wpt   = (u16*)take((size_t)GPT_L * GPT_D * GPT_D * 2);
  u16* w1t   = (u16*)take((size_t)GPT_L * GPT_FF * GPT_D * 2);
  u16* w2t   = (u16*)take((size_t)GPT_L * GPT_D * GPT_FF * 2);
  u16* wlmt  = (u16*)take((size_t)GPT_VP * GPT_D * 2);

  // weight conversions (fp32 -> bf16, pre-transposed to [N,K])
  {
    long long tq = (long long)GPT_L * GPT_D * GPT_D;
    int gq = (int)((tq + 255) / 256);
    conv_qkv_kernel<<<gq, 256, 0, stream>>>(Wq, wqt);
    conv_qkv_kernel<<<gq, 256, 0, stream>>>(Wk, wkt);
    conv_qkv_kernel<<<gq, 256, 0, stream>>>(Wv, wvt);
    conv_plain_kernel<<<gq, 256, 0, stream>>>(Wproj, wpt, GPT_D, GPT_D, GPT_D, tq);
    long long t1 = (long long)GPT_L * GPT_FF * GPT_D;
    int g1 = (int)((t1 + 255) / 256);
    conv_plain_kernel<<<g1, 256, 0, stream>>>(W1, w1t, GPT_D, GPT_FF, GPT_FF, t1);
    conv_plain_kernel<<<g1, 256, 0, stream>>>(W2, w2t, GPT_FF, GPT_D, GPT_D, t1);
    long long tl = (long long)GPT_VP * GPT_D;
    int gl = (int)((tl + 255) / 256);
    conv_plain_kernel<<<gl, 256, 0, stream>>>(Wlm, wlmt, GPT_D, GPT_V, GPT_VP, tl);
  }

  embed_kernel<<<(GPT_BT * GPT_D + 255) / 256, 256, 0, stream>>>(index, tok, pos, x);

  const dim3 gN512(GPT_D / BN, GPT_BT / BM);       // (4, 32)
  const dim3 gN2048(GPT_FF / BN, GPT_BT / BM);     // (16, 32)
  const dim3 gNlm(GPT_VP / BN, GPT_BT / BM);       // (393, 32)
  const int nvtrans = (GPT_B * GPT_H * GPT_HS * GPT_T + 255) / 256;

  for (int l = 0; l < GPT_L; ++l) {
    ln_kernel<<<GPT_BT, 256, 0, stream>>>(x, ln1g + l * GPT_D, ln1b + l * GPT_D, hb);
    gemm_bf16_kernel<<<gN512, 256, 0, stream>>>(hb, wqt + (size_t)l * GPT_D * GPT_D,
                                                nullptr, qb, GPT_D, GPT_D, GPT_D, 0);
    gemm_bf16_kernel<<<gN512, 256, 0, stream>>>(hb, wkt + (size_t)l * GPT_D * GPT_D,
                                                nullptr, kb, GPT_D, GPT_D, GPT_D, 0);
    gemm_bf16_kernel<<<gN512, 256, 0, stream>>>(hb, wvt + (size_t)l * GPT_D * GPT_D,
                                                nullptr, vb, GPT_D, GPT_D, GPT_D, 0);
    vtrans_kernel<<<nvtrans, 256, 0, stream>>>(vb, vtb);
    attn_kernel<<<dim3(GPT_T / 64, GPT_H, GPT_B), 128, 0, stream>>>(qb, kb, vtb, ab);
    gemm_bf16_kernel<<<gN512, 256, 0, stream>>>(ab, wpt + (size_t)l * GPT_D * GPT_D,
                                                bproj + l * GPT_D, x, GPT_D, GPT_D,
                                                GPT_D, 2);
    ln_kernel<<<GPT_BT, 256, 0, stream>>>(x, ln2g + l * GPT_D, ln2b + l * GPT_D, hb);
    gemm_bf16_kernel<<<gN2048, 256, 0, stream>>>(hb, w1t + (size_t)l * GPT_FF * GPT_D,
                                                 b1 + l * GPT_FF, ffm, GPT_D, GPT_FF,
                                                 GPT_FF, 1);
    gemm_bf16_kernel<<<gN512, 256, 0, stream>>>(ffm, w2t + (size_t)l * GPT_D * GPT_FF,
                                                b2 + l * GPT_D, x, GPT_FF, GPT_D,
                                                GPT_D, 2);
  }

  ln_kernel<<<GPT_BT, 256, 0, stream>>>(x, lnfg, lnfb, hb);
  gemm_bf16_kernel<<<gNlm, 256, 0, stream>>>(hb, wlmt, nullptr, out, GPT_D, GPT_V,
                                             GPT_V, 3);
}
